// TriangleAttention_21766894256697
// MI455X (gfx1250) — matrix operand order repaired and checked by reading, not compiled, not run on hardware
//
#include <hip/hip_runtime.h>
#include <hip/hip_bf16.h>

typedef __attribute__((ext_vector_type(16))) _Float16 v16h;
typedef __attribute__((ext_vector_type(2)))  _Float16 half2v;
typedef __attribute__((ext_vector_type(8)))  float    v8f;

#define S_LEN  384
#define D_DIM  128
#define H_N    4
#define HD_DIM 32
#define NPOS   (S_LEN * S_LEN)   // 147456

// ---------------------------------------------------------------------------
// WMMA fragment helpers (contiguous k = 16*(L/16) + s order in both A and B, wave32)
// ---------------------------------------------------------------------------

// A fragment: 16x32 f16 tile, row-major source, src points at (row0, k0).
// lane L: M = L%16 ; VGPR v holds K = 16*(L>=16) + 2*v + {0,1}
__device__ __forceinline__ v16h load_a_frag16x32(const _Float16* __restrict__ src,
                                                 int ld, int lane) {
  const int m   = lane & 15;
  const int khi = (lane >> 4) << 4;
  const _Float16* p = src + (size_t)m * ld;
  v16h a;
#pragma unroll
  for (int v = 0; v < 8; ++v) {
    const int kb = khi + (v << 1);
    a[2 * v]     = p[kb];
    a[2 * v + 1] = p[kb + 1];
  }
  return a;
}

// B fragment from a transposed source: B(k,n) = src[n*ld + k], src at (n0, k0).
// Per-lane reads are 16 contiguous f16 (32 bytes) -> vectorizes.
__device__ __forceinline__ v16h load_b_frag32x16_transposed(const _Float16* __restrict__ src,
                                                            int ld, int lane) {
  const int n = lane & 15;
  const _Float16* p = src + (size_t)n * ld + ((lane >> 4) << 4);
  v16h b;
#pragma unroll
  for (int e = 0; e < 16; ++e) b[e] = p[e];
  return b;
}

__device__ __forceinline__ v8f wmma_f32_f16(v16h a, v16h b, v8f c) {
  return __builtin_amdgcn_wmma_f32_16x16x32_f16(false, a, false, b, (short)0, c,
                                                false, false);
}

// CDNA5 async global -> LDS copy (tracked on ASYNCcnt).
__device__ __forceinline__ void async_b128_to_lds(unsigned int lds_off,
                                                  const void* gaddr) {
  asm volatile("global_load_async_to_lds_b128 %0, %1, off"
               :: "v"(lds_off), "v"(gaddr) : "memory");
}
__device__ __forceinline__ void wait_asynccnt0() {
  asm volatile("s_wait_asynccnt 0x0" ::: "memory");
}

// ---------------------------------------------------------------------------
// K0: convert weights f32 -> f16   [qkv_w 49152][g_w 16384][o_w 16384]
// ---------------------------------------------------------------------------
__global__ void k_convert_weights(const float* __restrict__ qkv_w,
                                  const float* __restrict__ g_w,
                                  const float* __restrict__ o_w,
                                  _Float16* __restrict__ dst) {
  const int idx = blockIdx.x * 256 + threadIdx.x;
  if (idx < 49152)       dst[idx] = (_Float16)qkv_w[idx];
  else if (idx < 65536)  dst[idx] = (_Float16)g_w[idx - 49152];
  else if (idx < 81920)  dst[idx] = (_Float16)o_w[idx - 65536];
}

// ---------------------------------------------------------------------------
// K1: LayerNorm + triangle-bias projection. One wave per position.
// ---------------------------------------------------------------------------
__global__ void k_layernorm_bias(const float* __restrict__ x,
                                 const float* __restrict__ ln_w,
                                 const float* __restrict__ ln_b,
                                 const float* __restrict__ bias_w,
                                 _Float16* __restrict__ ln16,
                                 float* __restrict__ bias_out) {
  const int lane = threadIdx.x & 31;
  const int wave = threadIdx.x >> 5;
  const int pos  = blockIdx.x * 8 + wave;

  const float4 xv = ((const float4*)(x + (size_t)pos * D_DIM))[lane];
  float xa[4] = {xv.x, xv.y, xv.z, xv.w};

  float s1 = xa[0] + xa[1] + xa[2] + xa[3];
  float s2 = xa[0]*xa[0] + xa[1]*xa[1] + xa[2]*xa[2] + xa[3]*xa[3];
#pragma unroll
  for (int off = 16; off >= 1; off >>= 1) {
    s1 += __shfl_xor(s1, off, 32);
    s2 += __shfl_xor(s2, off, 32);
  }
  const float mean = s1 * (1.0f / D_DIM);
  const float var  = s2 * (1.0f / D_DIM) - mean * mean;
  const float rstd = rsqrtf(var + 1e-5f);

  const int d0 = lane << 2;
  float lnv[4];
#pragma unroll
  for (int t = 0; t < 4; ++t) {
    const float v = (xa[t] - mean) * rstd * ln_w[d0 + t] + ln_b[d0 + t];
    lnv[t] = v;
    ln16[(size_t)pos * D_DIM + d0 + t] = (_Float16)v;
  }

  const int i = pos / S_LEN, j = pos - i * S_LEN;
#pragma unroll
  for (int h = 0; h < H_N; ++h) {
    float bsum = 0.0f;
#pragma unroll
    for (int t = 0; t < 4; ++t) bsum += lnv[t] * bias_w[h * D_DIM + d0 + t];
#pragma unroll
    for (int off = 16; off >= 1; off >>= 1) bsum += __shfl_xor(bsum, off, 32);
    if (lane == 0) bias_out[((size_t)h * S_LEN + i) * S_LEN + j] = bsum;
  }
}

// ---------------------------------------------------------------------------
// K2: fused qkv+g projection GEMM [N x 128] * [128 x 512] via WMMA.
// q/k scattered into [b][h][s][hd]; V stored TRANSPOSED [b][h][hd][s];
// g stored [pos][128] f16.
// ---------------------------------------------------------------------------
__global__ void k_proj_qkvg(const _Float16* __restrict__ ln16,
                            const _Float16* __restrict__ wqkv,  // [384][128] f16
                            const _Float16* __restrict__ wg,    // [128][128] f16
                            _Float16* __restrict__ qb, _Float16* __restrict__ kb,
                            _Float16* __restrict__ vt, _Float16* __restrict__ gb) {
  const int lane = threadIdx.x & 31;
  const int wave = threadIdx.x >> 5;
  const int row0 = blockIdx.x << 4;
  const int ct   = blockIdx.y * 8 + wave;  // col tile 0..31
  const int n0   = ct << 4;                // global col 0..511

  const _Float16* wsrc = (n0 < 384) ? (wqkv + (size_t)n0 * D_DIM)
                                    : (wg + (size_t)(n0 - 384) * D_DIM);
  v8f acc = {};
#pragma unroll
  for (int kc = 0; kc < 4; ++kc) {
    const v16h a = load_a_frag16x32(ln16 + (size_t)row0 * D_DIM + kc * 32, D_DIM, lane);
    const v16h b = load_b_frag32x16_transposed(wsrc + kc * 32, D_DIM, lane);
    acc = wmma_f32_f16(a, b, acc);
  }

  const int n  = lane & 15;
  const int mh = (lane >> 4) << 3;
#pragma unroll
  for (int r = 0; r < 8; ++r) {
    const int p = row0 + r + mh;
    const int c = n0 + n;
    const _Float16 val = (_Float16)acc[r];
    if (c < 384) {
      const int part = c >> 7;       // 0=q 1=k 2=v
      const int w = c & 127;
      const int h = w >> 5, d = w & 31;
      const int bidx = p / S_LEN, jj = p - bidx * S_LEN;
      const size_t head = (size_t)bidx * H_N + h;
      if (part == 0)      qb[(head * S_LEN + jj) * HD_DIM + d] = val;
      else if (part == 1) kb[(head * S_LEN + jj) * HD_DIM + d] = val;
      else                vt[(head * HD_DIM + d) * S_LEN + jj] = val;   // transposed
    } else {
      gb[(size_t)p * D_DIM + (c - 384)] = val;
    }
  }
}

// ---------------------------------------------------------------------------
// K3: attention. One wave per (b, h, q-tile of 16).
//   V^T tile DMA'd to LDS asynchronously (overlaps the score phase),
//   scores (16x384) -> LDS, vectorized softmax, P@V via WMMA, sigmoid gate.
// ---------------------------------------------------------------------------
__global__ void __launch_bounds__(32) k_attention(
    const _Float16* __restrict__ qb, const _Float16* __restrict__ kb,
    const _Float16* __restrict__ vt, const _Float16* __restrict__ gb,
    const float* __restrict__ bias, const float* __restrict__ mask,
    _Float16* __restrict__ og) {
  __shared__ float    sS[16][S_LEN + 4];       // 24832 B
  __shared__ _Float16 sP[16][S_LEN + 8];       // 12544 B
  __shared__ _Float16 sV[HD_DIM][S_LEN + 8];   // 25088 B  (total 62464 B)

  const int lane = threadIdx.x & 31;
  const int qt   = blockIdx.x;   // 0..23
  const int h    = blockIdx.y;   // 0..3
  const int bidx = blockIdx.z;   // 0..383
  const int q0   = qt << 4;

  const size_t head  = (size_t)bidx * H_N + h;
  const _Float16* qbase = qb + head * S_LEN * HD_DIM;
  const _Float16* kbase = kb + head * S_LEN * HD_DIM;
  const _Float16* vbase = vt + head * HD_DIM * S_LEN;   // [hd][s]
  const float scale = 0.17677669529663687f;             // 1/sqrt(HD)

  // --- kick off async DMA of the 32x384 f16 V^T tile into padded LDS rows ---
  {
    const unsigned int sv_base = (unsigned int)(uintptr_t)&sV[0][0];
    const char* vg = (const char*)vbase;
    for (int i = 0; i < 48; ++i) {                 // 48*32 lanes * 16B = 24576B
      const int chunk = i * 32 + lane;
      const int row   = chunk / 48;                // 0..31 (hd)
      const int c16   = chunk - row * 48;          // 0..47 (16B units along s)
      const unsigned int loff = sv_base + row * ((S_LEN + 8) * 2) + c16 * 16;
      const void* ga = vg + row * (S_LEN * 2) + c16 * 16;
      async_b128_to_lds(loff, ga);
    }
  }

  const int n  = lane & 15;
  const int mh = (lane >> 4) << 3;

  // --- scores: Q(16x32) @ K^T tiles, fused scale + bias + mask ---
  const v16h aq = load_a_frag16x32(qbase + (size_t)q0 * HD_DIM, HD_DIM, lane);
  for (int kt = 0; kt < S_LEN / 16; ++kt) {
    const v16h bk =
        load_b_frag32x16_transposed(kbase + (size_t)(kt << 4) * HD_DIM, HD_DIM, lane);
    v8f sc = {};
    sc = wmma_f32_f16(aq, bk, sc);
    const int kc = (kt << 4) + n;
#pragma unroll
    for (int r = 0; r < 8; ++r) {
      const int m = r + mh;
      sS[m][kc] = sc[r] * scale +
                  bias[((size_t)h * S_LEN + q0 + m) * S_LEN + kc] + mask[kc];
    }
  }
  __syncthreads();

  // --- softmax: each lane owns half of one row (float4 vectorized) ---
  {
    const int row = lane & 15;
    const int c0  = (lane >> 4) * (S_LEN / 2);
    float4* rp = (float4*)&sS[row][c0];            // 48 float4 per half-row
    float mx = -3.0e38f;
#pragma unroll 4
    for (int c = 0; c < 48; ++c) {
      const float4 t = rp[c];
      mx = fmaxf(fmaxf(fmaxf(mx, t.x), t.y), fmaxf(t.z, t.w));
    }
    mx = fmaxf(mx, __shfl_xor(mx, 16, 32));
    float sum = 0.0f;
#pragma unroll 2
    for (int c = 0; c < 48; ++c) {
      float4 t = rp[c];
      t.x = __expf(t.x - mx); t.y = __expf(t.y - mx);
      t.z = __expf(t.z - mx); t.w = __expf(t.w - mx);
      rp[c] = t;
      sum += (t.x + t.y) + (t.z + t.w);
    }
    sum += __shfl_xor(sum, 16, 32);
    const float inv = 1.0f / sum;
    half2v* pp = (half2v*)&sP[row][c0];
#pragma unroll 2
    for (int c = 0; c < 48; ++c) {
      const float4 t = rp[c];
      half2v lo = {(_Float16)(t.x * inv), (_Float16)(t.y * inv)};
      half2v hi = {(_Float16)(t.z * inv), (_Float16)(t.w * inv)};
      pp[2 * c]     = lo;
      pp[2 * c + 1] = hi;
    }
  }
  __syncthreads();
  wait_asynccnt0();     // V^T tile now resident in LDS

  // --- O = P(16x384) @ V(384x32), V read from LDS (contiguous per lane) ---
  v8f acc0 = {}, acc1 = {};
  for (int kc = 0; kc < S_LEN / 32; ++kc) {
    const v16h ap  = load_a_frag16x32(&sP[0][kc * 32], S_LEN + 8, lane);
    const v16h bv0 = load_b_frag32x16_transposed(&sV[0][kc * 32],  S_LEN + 8, lane);
    const v16h bv1 = load_b_frag32x16_transposed(&sV[16][kc * 32], S_LEN + 8, lane);
    acc0 = wmma_f32_f16(ap, bv0, acc0);
    acc1 = wmma_f32_f16(ap, bv1, acc1);
  }

  // --- sigmoid(g) gate, store gated output f16 ---
#pragma unroll
  for (int r = 0; r < 8; ++r) {
    const int m = r + mh;
    const size_t p = (size_t)bidx * S_LEN + q0 + m;
#pragma unroll
    for (int half = 0; half < 2; ++half) {
      const int d = half * 16 + n;
      const float gv = (float)gb[p * D_DIM + h * HD_DIM + d];
      const float ov = (half ? acc1[r] : acc0[r]) * (1.0f / (1.0f + __expf(-gv)));
      og[p * D_DIM + h * HD_DIM + d] = (_Float16)ov;
    }
  }
}

// ---------------------------------------------------------------------------
// K4: output projection GEMM [N x 128] * [128 x 128] -> f32 d_out
// ---------------------------------------------------------------------------
__global__ void k_out_proj(const _Float16* __restrict__ og,
                           const _Float16* __restrict__ wo,  // o_w [128][128] f16
                           float* __restrict__ out) {
  const int lane = threadIdx.x & 31;
  const int wave = threadIdx.x >> 5;
  const int row0 = blockIdx.x << 4;
  const int n0   = wave << 4;  // 0..112

  v8f acc = {};
#pragma unroll
  for (int kc = 0; kc < 4; ++kc) {
    const v16h a = load_a_frag16x32(og + (size_t)row0 * D_DIM + kc * 32, D_DIM, lane);
    const v16h b =
        load_b_frag32x16_transposed(wo + (size_t)n0 * D_DIM + kc * 32, D_DIM, lane);
    acc = wmma_f32_f16(a, b, acc);
  }
  const int n  = lane & 15;
  const int mh = (lane >> 4) << 3;
#pragma unroll
  for (int r = 0; r < 8; ++r)
    out[(size_t)(row0 + r + mh) * D_DIM + n0 + n] = acc[r];
}

// ---------------------------------------------------------------------------
extern "C" void kernel_launch(void* const* d_in, const int* in_sizes, int n_in,
                              void* d_out, int out_size, void* d_ws, size_t ws_size,
                              hipStream_t stream) {
  (void)in_sizes; (void)n_in; (void)out_size; (void)ws_size;

  const float* x         = (const float*)d_in[0];
  const float* attn_mask = (const float*)d_in[1];
  const float* ln_w      = (const float*)d_in[2];
  const float* ln_b      = (const float*)d_in[3];
  const float* bias_w    = (const float*)d_in[4];
  const float* qkv_w     = (const float*)d_in[5];
  const float* g_w       = (const float*)d_in[6];
  const float* o_w       = (const float*)d_in[7];

  char* ws = (char*)d_ws;
  size_t off = 0;
  auto take = [&](size_t bytes) {
    char* p = ws + off;
    off += (bytes + 255) & ~(size_t)255;
    return p;
  };

  _Float16* ln16 = (_Float16*)take((size_t)NPOS * D_DIM * 2);  // reused as og
  _Float16* qb   = (_Float16*)take((size_t)NPOS * D_DIM * 2);
  _Float16* kb   = (_Float16*)take((size_t)NPOS * D_DIM * 2);
  _Float16* vt   = (_Float16*)take((size_t)NPOS * D_DIM * 2);
  _Float16* gb   = (_Float16*)take((size_t)NPOS * D_DIM * 2);
  float*    bias = (float*)take((size_t)H_N * NPOS * sizeof(float));
  _Float16* w16  = (_Float16*)take((size_t)81920 * 2);
  _Float16* wqkv16 = w16;
  _Float16* wg16   = w16 + 49152;
  _Float16* wo16   = w16 + 65536;

  k_convert_weights<<<320, 256, 0, stream>>>(qkv_w, g_w, o_w, w16);
  k_layernorm_bias<<<NPOS / 8, 256, 0, stream>>>(x, ln_w, ln_b, bias_w, ln16, bias);
  k_proj_qkvg<<<dim3(NPOS / 16, 4), 256, 0, stream>>>(ln16, wqkv16, wg16,
                                                      qb, kb, vt, gb);
  k_attention<<<dim3(S_LEN / 16, H_N, S_LEN), 32, 0, stream>>>(
      qb, kb, vt, gb, bias, attn_mask, ln16 /* og reuses ln buffer */);
  k_out_proj<<<NPOS / 16, 256, 0, stream>>>(ln16, wo16, (float*)d_out);
}
